// VisionDynamicSparseAttention_13932873908467
// MI455X (gfx1250) — matrix operand order repaired and checked by reading, not compiled, not run on hardware
//
#include <hip/hip_runtime.h>
#include <hip/hip_bf16.h>

typedef __attribute__((ext_vector_type(16))) _Float16 v16h;
typedef __attribute__((ext_vector_type(8)))  _Float16 v8h;
typedef __attribute__((ext_vector_type(4)))  _Float16 v4h;
typedef __attribute__((ext_vector_type(8)))  float    v8f;

#define Bsz 4
#define Lq  1024
#define Emb 1024
#define NH  16
#define Dh  64
#define KEEP_K 409   // max(1, int(1024*0.4))

__device__ __forceinline__ v16h cat8(v8h lo, v8h hi) {
    return __builtin_shufflevector(lo, hi, 0,1,2,3,4,5,6,7,8,9,10,11,12,13,14,15);
}
__device__ __forceinline__ v8f wmma_f16(v16h a, v16h b, v8f c) {
    return __builtin_amdgcn_wmma_f32_16x16x32_f16(false, a, false, b, (short)0, c, false, false);
}

// ---------------------------------------------------------------------------
// gfx1250 async-copy helpers (ASYNCcnt path)
// ---------------------------------------------------------------------------
__device__ __forceinline__ void async_b128(unsigned int lds_off, const void* gptr)
{
    asm volatile("global_load_async_to_lds_b128 %0, %1, off"
                 :: "v"(lds_off), "v"(gptr) : "memory");
}
__device__ __forceinline__ void wait_async0()
{
    asm volatile("s_wait_asynccnt 0x0" ::: "memory");
}

// ---------------------------------------------------------------------------
// Kernel 0: bulk fp32 -> f16 conversion (hoisted out of GEMM hot loops).
// ---------------------------------------------------------------------------
__global__ __launch_bounds__(256)
void cvt_f16_kernel(const float* __restrict__ src, _Float16* __restrict__ dst, int n)
{
    const int i = (blockIdx.x * 256 + threadIdx.x) * 4;
    if (i < n) {
        const float4 v = *(const float4*)(src + i);
        v4h o = { (_Float16)v.x, (_Float16)v.y, (_Float16)v.z, (_Float16)v.w };
        *(v4h*)(dst + i) = o;
    }
}

// ---------------------------------------------------------------------------
// Kernel 1: QKV projection, f16 WMMA, W tile async-staged in LDS.
// q,k stored [B,H,L,D]; v stored TRANSPOSED [B,H,D,L].
// Block = 128 threads (4 waves); wave owns a 16x64 output tile; K-step 64.
// ---------------------------------------------------------------------------
__global__ __launch_bounds__(128)
void qkv_gemm_kernel(const _Float16* __restrict__ X16, const _Float16* __restrict__ W16,
                     const float* __restrict__ bias,
                     _Float16* __restrict__ q16, _Float16* __restrict__ k16,
                     _Float16* __restrict__ vT16)
{
    __shared__ __align__(16) _Float16 blds[2][64 * 64];   // [n][k], 8KB per buffer

    const int Kdim = Emb;
    const int tid  = threadIdx.x;
    const int lane = tid & 31;
    const int wid  = tid >> 5;
    const int m0   = blockIdx.x * 64 + wid * 16;
    const int n0   = blockIdx.y * 64;
    const int row  = lane & 15;
    const int koff = (lane & 16) ? 8  : 0;
    const int kb0  = (lane & 16) ? 16 : 0;
    const int hi8  = (lane & 16) ? 8  : 0;

    const unsigned int b_lds[2] = { (unsigned int)(size_t)&blds[0][0],
                                    (unsigned int)(size_t)&blds[1][0] };
    // staging: 64 rows x 128B; 128 threads x 4 x 16B chunks
    const int srow    = tid >> 1;
    const int schunk0 = (tid & 1) * 4;
    const _Float16* wbase = W16 + (size_t)n0 * Kdim;

    auto issue_b = [&](int kb, int buf) {
#pragma unroll
        for (int c = 0; c < 4; ++c) {
            async_b128(b_lds[buf] + (unsigned int)((srow * 64 + (schunk0 + c) * 8) * 2),
                       wbase + (size_t)srow * Kdim + kb + (schunk0 + c) * 8);
        }
    };

    const _Float16* ap = X16 + (size_t)(m0 + row) * Kdim;
    v8f acc[4] = {{}, {}, {}, {}};

    issue_b(0, 0);
    for (int it = 0; it < Kdim / 64; ++it) {
        const int kb = it * 64;
        const _Float16* bt_base = &blds[it & 1][0];
        wait_async0();
        __syncthreads();
        if (it + 1 < Kdim / 64) issue_b(kb + 64, (it + 1) & 1);

#pragma unroll
        for (int kk = 0; kk < 2; ++kk) {
            const _Float16* arow = ap + kb + kk * 32;
            const v16h a = cat8(*(const v8h*)(arow + kb0),
                                *(const v8h*)(arow + kb0 + 8));
#pragma unroll
            for (int j = 0; j < 4; ++j) {
                const _Float16* bp = bt_base + (16 * j + row) * 64 + kk * 32 + kb0;
                const v16h bt = cat8(*(const v8h*)bp, *(const v8h*)(bp + 8));
                acc[j] = wmma_f16(a, bt, acc[j]);
            }
        }
        if (it + 2 < Kdim / 64) __builtin_prefetch(ap + kb + 128, 0, 3);
    }

    const int t = n0 >> 10;            // 0=q 1=k 2=v (n0 is 64-aligned)
    const int h = (n0 >> 6) & 15;

#pragma unroll
    for (int j = 0; j < 4; ++j) {
        const int n  = n0 + 16 * j + row;   // C col = lane&15
        const float bv = bias[n];
        const int d = n & 63;
#pragma unroll
        for (int g = 0; g < 8; ++g) {
            const int mrow = m0 + g + hi8;
            const int bb = mrow >> 10, ll = mrow & 1023;
            const _Float16 val = (_Float16)(acc[j][g] + bv);
            if (t == 2) {   // uniform branch per block
                vT16[((size_t)(bb * NH + h) * Dh + d) * Lq + ll] = val;
            } else {
                _Float16* dst = (t == 0) ? q16 : k16;
                dst[(((size_t)(bb * NH + h) * Lq) + ll) * Dh + d] = val;
            }
        }
    }
}

// ---------------------------------------------------------------------------
// Kernel 2: routing layer 1. h1 = relu(X16 * r_w1^T + b1), fp32 [4096, 256].
// Same structure as kernel 1.
// ---------------------------------------------------------------------------
__global__ __launch_bounds__(128)
void routing_gemm_kernel(const _Float16* __restrict__ X16, const _Float16* __restrict__ W16,
                         const float* __restrict__ bias, float* __restrict__ h1)
{
    __shared__ __align__(16) _Float16 blds[2][64 * 64];

    const int Kdim = Emb;
    const int tid  = threadIdx.x;
    const int lane = tid & 31;
    const int wid  = tid >> 5;
    const int m0   = blockIdx.x * 64 + wid * 16;
    const int n0   = blockIdx.y * 64;
    const int row  = lane & 15;
    const int koff = (lane & 16) ? 8  : 0;
    const int kb0  = (lane & 16) ? 16 : 0;
    const int hi8  = (lane & 16) ? 8  : 0;

    const unsigned int b_lds[2] = { (unsigned int)(size_t)&blds[0][0],
                                    (unsigned int)(size_t)&blds[1][0] };
    const int srow    = tid >> 1;
    const int schunk0 = (tid & 1) * 4;
    const _Float16* wbase = W16 + (size_t)n0 * Kdim;

    auto issue_b = [&](int kb, int buf) {
#pragma unroll
        for (int c = 0; c < 4; ++c) {
            async_b128(b_lds[buf] + (unsigned int)((srow * 64 + (schunk0 + c) * 8) * 2),
                       wbase + (size_t)srow * Kdim + kb + (schunk0 + c) * 8);
        }
    };

    const _Float16* ap = X16 + (size_t)(m0 + row) * Kdim;
    v8f acc[4] = {{}, {}, {}, {}};

    issue_b(0, 0);
    for (int it = 0; it < Kdim / 64; ++it) {
        const int kb = it * 64;
        const _Float16* bt_base = &blds[it & 1][0];
        wait_async0();
        __syncthreads();
        if (it + 1 < Kdim / 64) issue_b(kb + 64, (it + 1) & 1);

#pragma unroll
        for (int kk = 0; kk < 2; ++kk) {
            const _Float16* arow = ap + kb + kk * 32;
            const v16h a = cat8(*(const v8h*)(arow + kb0),
                                *(const v8h*)(arow + kb0 + 8));
#pragma unroll
            for (int j = 0; j < 4; ++j) {
                const _Float16* bp = bt_base + (16 * j + row) * 64 + kk * 32 + kb0;
                const v16h bt = cat8(*(const v8h*)bp, *(const v8h*)(bp + 8));
                acc[j] = wmma_f16(a, bt, acc[j]);
            }
        }
        if (it + 2 < Kdim / 64) __builtin_prefetch(ap + kb + 128, 0, 3);
    }

#pragma unroll
    for (int j = 0; j < 4; ++j) {
        const int n  = n0 + 16 * j + row;
        const float bv = bias[n];
#pragma unroll
        for (int g = 0; g < 8; ++g) {
            const int mrow = m0 + g + hi8;
            float v = acc[j][g] + bv;
            h1[(size_t)mrow * 256 + n] = v > 0.f ? v : 0.f;
        }
    }
}

// ---------------------------------------------------------------------------
// Kernel 3: routing layer 2 + double sigmoid -> scores [B,H,L] fp32.
// ---------------------------------------------------------------------------
__global__ __launch_bounds__(256)
void routing2_kernel(const float* __restrict__ h1, const float* __restrict__ w2,
                     const float* __restrict__ b2, float* __restrict__ scores)
{
    const int tid = threadIdx.x;
    const int tok = blockIdx.x * 16 + (tid >> 4);
    const int h   = tid & 15;
    const float* hp = h1 + (size_t)tok * 256;
    const float* wp = w2 + (size_t)h * 256;
    float acc = b2[h];
#pragma unroll 8
    for (int i = 0; i < 256; ++i) acc += hp[i] * wp[i];
    float s1 = 1.f / (1.f + __expf(-acc));
    float s2 = 1.f / (1.f + __expf(-s1));
    const int b = tok >> 10, l = tok & 1023;
    scores[((size_t)(b * NH + h)) * Lq + l] = s2;
}

// ---------------------------------------------------------------------------
// Kernel 4: exact top-k keep mask per (b,h) via rank counting.
// ---------------------------------------------------------------------------
__global__ __launch_bounds__(1024)
void topk_mask_kernel(const float* __restrict__ scores, unsigned char* __restrict__ keep)
{
    __shared__ float s[Lq];
    const int bh = blockIdx.x;
    const int i = threadIdx.x;
    s[i] = scores[(size_t)bh * Lq + i];
    __syncthreads();
    const float v = s[i];
    int rank = 0;
    for (int j = 0; j < Lq; ++j) {
        const float sj = s[j];
        rank += (sj > v) || ((sj == v) && (j < i));
    }
    keep[(size_t)bh * Lq + i] = (rank < KEEP_K) ? (unsigned char)1 : (unsigned char)0;
}

// ---------------------------------------------------------------------------
// Kernel 5: masked flash attention. Grid (8, B*H), block 256 (8 waves).
// K/V kv-tiles (32 wide) async-staged in LDS, double-buffered, shared by all
// 8 waves. Wave owns a 16-row Q tile.
// ---------------------------------------------------------------------------
__global__ __launch_bounds__(256)
void attention_kernel(const _Float16* __restrict__ q16, const _Float16* __restrict__ k16,
                      const _Float16* __restrict__ vT16, const unsigned char* __restrict__ keep,
                      float* __restrict__ out)
{
    __shared__ __align__(16) _Float16 klds[2][32 * 64];   // [kv][d]
    __shared__ __align__(16) _Float16 vlds[2][64 * 32];   // [d][kv]
    __shared__ __align__(16) _Float16 plds[8][16 * 32];   // per-wave P staging

    const int tid  = threadIdx.x;
    const int lane = tid & 31;
    const int wid  = tid >> 5;
    const int bh   = blockIdx.y;
    const int b = bh >> 4, h = bh & 15;
    const int q0   = (blockIdx.x * 8 + wid) * 16;
    const int row  = lane & 15;
    const int koff = (lane & 16) ? 8  : 0;
    const int kb0  = (lane & 16) ? 16 : 0;
    const int hi8  = (lane & 16) ? 8  : 0;

    const _Float16* qb  = q16  + (size_t)bh * Lq * Dh;
    const _Float16* kbp = k16  + (size_t)bh * Lq * Dh;
    const _Float16* vtp = vT16 + (size_t)bh * Dh * Lq;
    const unsigned char* kp = keep + (size_t)bh * Lq;

    const unsigned int kb_lds[2] = { (unsigned int)(size_t)&klds[0][0],
                                     (unsigned int)(size_t)&klds[1][0] };
    const unsigned int vb_lds[2] = { (unsigned int)(size_t)&vlds[0][0],
                                     (unsigned int)(size_t)&vlds[1][0] };

    const int kvrow = tid >> 3, kchunk = tid & 7;   // K: 8 x 16B per kv row
    const int drow  = tid >> 2, vchunk = tid & 3;   // V: 4 x 16B per d row

    auto issue_tile = [&](int kv0, int buf) {
        async_b128(kb_lds[buf] + (unsigned int)((kvrow * 64 + kchunk * 8) * 2),
                   kbp + (size_t)(kv0 + kvrow) * Dh + kchunk * 8);
        async_b128(vb_lds[buf] + (unsigned int)((drow * 32 + vchunk * 8) * 2),
                   vtp + (size_t)drow * Lq + kv0 + vchunk * 8);
    };

    // Q tile in A layout: two K-steps across D=64.
    const _Float16* qr = qb + (size_t)(q0 + row) * Dh;
    const v16h qa0 = cat8(*(const v8h*)(qr + kb0),      *(const v8h*)(qr + kb0 + 8));
    const v16h qa1 = cat8(*(const v8h*)(qr + 32 + kb0), *(const v8h*)(qr + 40 + kb0));

    v8f o0 = {}, o1 = {}, o2 = {}, o3 = {};
    float m[8], l[8];
#pragma unroll
    for (int g = 0; g < 8; ++g) { m[g] = -1e30f; l[g] = 0.f; }
    const float scale = 0.125f;   // 1/sqrt(64)

    issue_tile(0, 0);

    for (int it = 0; it < Lq / 32; ++it) {
        const int kv0 = it * 32;
        const _Float16* kt = &klds[it & 1][0];
        const _Float16* vt = &vlds[it & 1][0];

        wait_async0();        // my async copies landed
        __syncthreads();      // everyone's landed; previous buffer fully consumed
        if (it + 1 < Lq / 32) issue_tile(kv0 + 32, (it + 1) & 1);

        // ---- S = Q K^T, two 16-wide N sub-tiles, K from LDS ----
        v8f s[2];
#pragma unroll
        for (int j = 0; j < 2; ++j) {
            const _Float16* kr = kt + (16 * j + row) * 64;
            const v16h b0 = cat8(*(const v8h*)(kr + kb0),      *(const v8h*)(kr + kb0 + 8));
            const v16h b1 = cat8(*(const v8h*)(kr + 32 + kb0), *(const v8h*)(kr + 40 + kb0));
            v8f sj = {};
            sj = wmma_f16(qa0, b0, sj);
            sj = wmma_f16(qa1, b1, sj);
            s[j] = sj;
        }

        const float kf0 = kp[kv0 + row]      ? 1.f : 0.f;
        const float kf1 = kp[kv0 + 16 + row] ? 1.f : 0.f;

        // ---- online softmax (row = g + hi8, col = lane&15) ----
        float alpha[8];
#pragma unroll
        for (int g = 0; g < 8; ++g) {
            float s0v = (kf0 != 0.f) ? s[0][g] * scale : -1e30f;
            float s1v = (kf1 != 0.f) ? s[1][g] * scale : -1e30f;
            float rmax = fmaxf(s0v, s1v);
#pragma unroll
            for (int d = 1; d < 16; d <<= 1) rmax = fmaxf(rmax, __shfl_xor(rmax, d, 32));
            const float mnew = fmaxf(m[g], rmax);
            const float a = __expf(m[g] - mnew);
            const float p0 = __expf(s0v - mnew) * kf0;
            const float p1 = __expf(s1v - mnew) * kf1;
            float ps = p0 + p1;
#pragma unroll
            for (int d = 1; d < 16; d <<= 1) ps += __shfl_xor(ps, d, 32);
            l[g] = l[g] * a + ps;
            m[g] = mnew;
            alpha[g] = a;
            const int r = g + hi8;
            plds[wid][r * 32 + row]      = (_Float16)p0;
            plds[wid][r * 32 + 16 + row] = (_Float16)p1;
        }

#pragma unroll
        for (int g = 0; g < 8; ++g) {
            o0[g] *= alpha[g]; o1[g] *= alpha[g];
            o2[g] *= alpha[g]; o3[g] *= alpha[g];
        }

        // Same-wave DS ops are in-order; barrier only fences scheduling.
        __builtin_amdgcn_wave_barrier();
        const _Float16* prow = &plds[wid][row * 32];
        const v16h pa = cat8(*(const v8h*)(prow + kb0), *(const v8h*)(prow + kb0 + 8));
        __builtin_amdgcn_wave_barrier();

        // ---- O += P V over 4 D sub-tiles (V^T from LDS, contiguous) ----
#pragma unroll
        for (int t = 0; t < 4; ++t) {
            const _Float16* vr = vt + (16 * t + row) * 32;
            const v16h bt = cat8(*(const v8h*)(vr + kb0), *(const v8h*)(vr + kb0 + 8));
            v8f& ot = (t == 0) ? o0 : (t == 1) ? o1 : (t == 2) ? o2 : o3;
            ot = wmma_f16(pa, bt, ot);
        }
    }

    // ---- normalize and store out[b][l][h*64 + d] ----
    float* ob = out + (size_t)b * Lq * Emb + (size_t)h * Dh;
#pragma unroll
    for (int g = 0; g < 8; ++g) {
        const float inv = 1.f / l[g];
        float* orow = ob + (size_t)(q0 + g + hi8) * Emb;
        orow[ 0 + row] = o0[g] * inv;
        orow[16 + row] = o1[g] * inv;
        orow[32 + row] = o2[g] * inv;
        orow[48 + row] = o3[g] * inv;
    }
}

// ---------------------------------------------------------------------------
extern "C" void kernel_launch(void* const* d_in, const int* in_sizes, int n_in,
                              void* d_out, int out_size, void* d_ws, size_t ws_size,
                              hipStream_t stream)
{
    const float* hidden = (const float*)d_in[0];
    const float* qkv_w  = (const float*)d_in[1];
    const float* qkv_b  = (const float*)d_in[2];
    const float* r_w1   = (const float*)d_in[3];
    const float* r_b1   = (const float*)d_in[4];
    const float* r_w2   = (const float*)d_in[5];
    const float* r_b2   = (const float*)d_in[6];
    float* out = (float*)d_out;

    char* ws = (char*)d_ws;
    size_t off = 0;
    auto alloc = [&](size_t bytes) -> void* {
        off = (off + 255) & ~(size_t)255;
        void* p = ws + off;
        off += bytes;
        return p;
    };
    const size_t qkv_elems = (size_t)Bsz * NH * Lq * Dh;     // 4,194,304
    const int n_x  = Bsz * Lq * Emb;                          // 4,194,304
    const int n_wq = 3 * Emb * Emb;                           // 3,145,728
    const int n_w1 = (Emb / 4) * Emb;                         // 262,144

    _Float16* q16   = (_Float16*)alloc(qkv_elems * sizeof(_Float16));
    _Float16* k16   = (_Float16*)alloc(qkv_elems * sizeof(_Float16));
    _Float16* vT16  = (_Float16*)alloc(qkv_elems * sizeof(_Float16));
    float* h1       = (float*)alloc((size_t)Bsz * Lq * 256 * sizeof(float));
    float* scores   = (float*)alloc((size_t)Bsz * NH * Lq * sizeof(float));
    unsigned char* keepm = (unsigned char*)alloc((size_t)Bsz * NH * Lq);
    _Float16* x16   = (_Float16*)alloc((size_t)n_x  * sizeof(_Float16));
    _Float16* wq16  = (_Float16*)alloc((size_t)n_wq * sizeof(_Float16));
    _Float16* w116  = (_Float16*)alloc((size_t)n_w1 * sizeof(_Float16));

    // 0) f32 -> f16 pre-conversion
    cvt_f16_kernel<<<n_x  / 1024, 256, 0, stream>>>(hidden, x16,  n_x);
    cvt_f16_kernel<<<n_wq / 1024, 256, 0, stream>>>(qkv_w,  wq16, n_wq);
    cvt_f16_kernel<<<n_w1 / 1024, 256, 0, stream>>>(r_w1,   w116, n_w1);

    // 1) QKV projection (M=4096, N=3072, K=1024)
    qkv_gemm_kernel<<<dim3(64, 48), 128, 0, stream>>>(x16, wq16, qkv_b, q16, k16, vT16);
    // 2) Routing layer 1 (M=4096, N=256, K=1024)
    routing_gemm_kernel<<<dim3(64, 4), 128, 0, stream>>>(x16, w116, r_b1, h1);
    // 3) Routing layer 2 + sigmoids -> [B,H,L]
    routing2_kernel<<<256, 256, 0, stream>>>(h1, r_w2, r_b2, scores);
    // 4) Top-409 keep mask per (b,h)
    topk_mask_kernel<<<Bsz * NH, 1024, 0, stream>>>(scores, keepm);
    // 5) Masked flash attention with async double-buffered K/V tiles
    attention_kernel<<<dim3(8, Bsz * NH), 256, 0, stream>>>(q16, k16, vT16, keepm, out);
}
